// CausalSelfAttention_57982058496181
// MI455X (gfx1250) — matrix operand order repaired and checked by reading, not compiled, not run on hardware
//
#include <hip/hip_runtime.h>
#include <hip/hip_bf16.h>

// ---------------------------------------------------------------------------
// Problem constants (from reference): B=2, T=2048, C=2048, H=16, D=128
// ---------------------------------------------------------------------------

typedef __bf16 bf16_t;
typedef __attribute__((ext_vector_type(16))) __bf16 v16bf;
typedef __attribute__((ext_vector_type(8)))  __bf16 v8bf;
typedef __attribute__((ext_vector_type(4)))  __bf16 v4bf;
typedef __attribute__((ext_vector_type(8)))  float  v8f;

#define WMMA_BF16(a, b, c) \
  __builtin_amdgcn_wmma_f32_16x16x32_bf16(false, (a), false, (b), (short)0, (c), false, false)

// A fragment (16x32 bf16, row-major source, leading dim `ld` in elements):
// lanes 0-15: row M=lane,  elements K = 0..15 contiguous
// lanes 16-31: row M=lane-16, elements K = 16..31 contiguous
static __device__ inline v16bf load_a_frag(const bf16_t* tile, int ld, int lane) {
  const int half = (lane >> 4) & 1;
  const int l = lane & 15;
  const bf16_t* row = tile + (size_t)l * ld;
  v8bf lo = *(const v8bf*)(row + half * 16);
  v8bf hi = *(const v8bf*)(row + half * 16 + 8);
  v16bf r;
#pragma unroll
  for (int i = 0; i < 8; ++i) { r[i] = lo[i]; r[i + 8] = hi[i]; }
  return r;
}

// B fragment (32x16, from an NxK row-major matrix, leading dim `ld`):
// lane n (0-15): column N=n, K = 0..15 contiguous; lanes 16-31: K = 16..31
static __device__ inline v16bf load_b_frag(const bf16_t* tile, int ld, int lane) {
  const int half = (lane >> 4) & 1;
  const int l = lane & 15;
  return *(const v16bf*)(tile + (size_t)l * ld + half * 16);
}

// Async 16B global -> LDS copy (per-lane), tracked by ASYNCcnt.
// lds_off: 32-bit LDS byte offset (low 32 bits of a flat shared pointer).
static __device__ inline void async_copy_b128(unsigned lds_off, const bf16_t* gaddr) {
  asm volatile("global_load_async_to_lds_b128 %0, %1, off"
               :: "v"(lds_off), "v"(gaddr) : "memory");
}

// ---------------------------------------------------------------------------
// 1) f32 -> bf16 conversion (vectorized x4; all sizes divisible by 4)
// ---------------------------------------------------------------------------
__global__ void k_cvt(const float* __restrict__ in, bf16_t* __restrict__ out, int n4) {
  int i = blockIdx.x * blockDim.x + threadIdx.x;
  if (i < n4) {
    float4 f = ((const float4*)in)[i];
    v4bf o;
    o[0] = (bf16_t)f.x; o[1] = (bf16_t)f.y; o[2] = (bf16_t)f.z; o[3] = (bf16_t)f.w;
    ((v4bf*)out)[i] = o;
  }
}

// ---------------------------------------------------------------------------
// Shared GEMM core: block = 8 waves, tile M=128 x N=128, K-step 32.
// B tile (128 rows x 32 K = 8 KB) staged to LDS with async copies, double
// buffered: stage buf^1 for k0+32, s_wait_asynccnt, barrier, compute from
// buf, barrier. A fragments stay direct global loads (per-wave unique).
// ---------------------------------------------------------------------------
struct GemmAcc { v8f acc[8]; };

static __device__ inline GemmAcc gemm_core_128(
    const bf16_t* __restrict__ A, const bf16_t* __restrict__ Wm,
    int K, int m0, int n0, bf16_t* ldsB /* [2][128*32] */) {
  const int tid = threadIdx.x;
  const int lane = tid & 31;

  // each thread copies 2 chunks of 16B (8 bf16): chunk c -> row j=c>>2, kpart=(c&3)*8
  const int c0 = tid, c1 = tid + 256;
  const int j0 = c0 >> 2, kp0 = (c0 & 3) * 8;
  const int j1 = c1 >> 2, kp1 = (c1 & 3) * 8;
  const bf16_t* g0 = Wm + (size_t)(n0 + j0) * K + kp0;
  const bf16_t* g1 = Wm + (size_t)(n0 + j1) * K + kp1;
  unsigned lo0[2], lo1[2];
#pragma unroll
  for (int b = 0; b < 2; ++b) {
    lo0[b] = (unsigned)(size_t)&ldsB[b * 4096 + j0 * 32 + kp0];
    lo1[b] = (unsigned)(size_t)&ldsB[b * 4096 + j1 * 32 + kp1];
  }

  GemmAcc r;
#pragma unroll
  for (int ct = 0; ct < 8; ++ct) r.acc[ct] = v8f{};

  const bf16_t* Ab = A + (size_t)m0 * K;
  const int half = lane >> 4, l = lane & 15;

  // prologue: stage buffer 0
  async_copy_b128(lo0[0], g0);
  async_copy_b128(lo1[0], g1);

  for (int k0 = 0; k0 < K; k0 += 32) {
    const int cur = (k0 >> 5) & 1;
    if (k0 + 32 < K) {
      async_copy_b128(lo0[cur ^ 1], g0 + k0 + 32);
      async_copy_b128(lo1[cur ^ 1], g1 + k0 + 32);
      asm volatile("s_wait_asynccnt 0x2" ::: "memory");   // current buffer done
    } else {
      asm volatile("s_wait_asynccnt 0x0" ::: "memory");
    }
    __syncthreads();                                      // tile visible to all waves

    v16bf a = load_a_frag(Ab + k0, K, lane);
    const bf16_t* bb = ldsB + cur * 4096;
    v16bf bfr[8];
#pragma unroll
    for (int ct = 0; ct < 8; ++ct)
      bfr[ct] = *(const v16bf*)&bb[(ct * 16 + l) * 32 + half * 16];
#pragma unroll
    for (int ct = 0; ct < 8; ++ct)
      r.acc[ct] = WMMA_BF16(a, bfr[ct], r.acc[ct]);

    __syncthreads();                                      // retire buffer before overwrite
  }
  return r;
}

// ---------------------------------------------------------------------------
// 2) QKV GEMM: out[n,j] = sum_c X[n,c] * Wqkv[j,c]
//    M=4096 (B*T), N=6144 (3C), K=2048.
//    Epilogue scatters into Q,K as [B,H,T,D] and V transposed as [B,H,D,T].
// ---------------------------------------------------------------------------
__global__ __launch_bounds__(256) void k_gemm_qkv(
    const bf16_t* __restrict__ X, const bf16_t* __restrict__ W,
    bf16_t* __restrict__ Qr, bf16_t* __restrict__ Kr, bf16_t* __restrict__ Vt) {
  constexpr int K = 2048, T = 2048, H = 16, D = 128;
  __shared__ __align__(64) bf16_t ldsB[2 * 128 * 32];

  const int lane = threadIdx.x & 31;
  const int wid  = threadIdx.x >> 5;
  const int m0 = blockIdx.y * 128 + wid * 16;
  const int n0 = blockIdx.x * 128;

  GemmAcc g = gemm_core_128(X, W, K, m0, n0, ldsB);

  const int half = lane >> 4, l = lane & 15;
  const int region = n0 >> 11;              // 0:q 1:k 2:v (128 | 2048, so uniform)
  const int bq = m0 >> 11;                  // batch index
  const int tlocal = (m0 & 2047) + half * 8;

  if (region < 2) {
    bf16_t* dst = (region == 0) ? Qr : Kr;
#pragma unroll
    for (int ct = 0; ct < 8; ++ct) {
      int j = (n0 & 2047) + ct * 16 + l;
      int h = j >> 7, d = j & 127;
      size_t base = (((size_t)bq * H + h) * T) * D + (size_t)d;
#pragma unroll
      for (int r = 0; r < 8; ++r)
        dst[base + (size_t)(tlocal + r) * D] = (bf16_t)g.acc[ct][r];
    }
  } else {
#pragma unroll
    for (int ct = 0; ct < 8; ++ct) {
      int j = (n0 & 2047) + ct * 16 + l;
      int h = j >> 7, d = j & 127;
      size_t base = (((size_t)bq * H + h) * D + d) * T + tlocal;
      v8bf pk;
#pragma unroll
      for (int r = 0; r < 8; ++r) pk[r] = (bf16_t)g.acc[ct][r];
      *(v8bf*)(Vt + base) = pk;             // 8 consecutive t -> one 16B store
    }
  }
}

// ---------------------------------------------------------------------------
// 3) RoPE in place on Q and K [B,H,T,D]; one thread per (d, d+64) pair.
// ---------------------------------------------------------------------------
__global__ void k_rope(bf16_t* __restrict__ Q, bf16_t* __restrict__ Kt, int total) {
  int idx = blockIdx.x * blockDim.x + threadIdx.x;
  if (idx >= total) return;
  const int i  = idx & 63;
  const int t  = (idx >> 6) & 2047;
  const int bh = idx >> 17;                 // / (64*2048)
  size_t base = ((size_t)bh * 2048 + t) * 128;
  // inv_freq[i] = 10000^(-i/64) ;  ln(10000) = 9.210340371976184
  float inv = __expf(-(float)i * (9.210340371976184f / 64.f));
  float ang = (float)t * inv;
  float s, c;
  __sincosf(ang, &s, &c);
  float q1 = (float)Q[base + i], q2 = (float)Q[base + i + 64];
  Q[base + i]      = (bf16_t)(q1 * c - q2 * s);
  Q[base + i + 64] = (bf16_t)(q2 * c + q1 * s);
  float k1 = (float)Kt[base + i], k2 = (float)Kt[base + i + 64];
  Kt[base + i]      = (bf16_t)(k1 * c - k2 * s);
  Kt[base + i + 64] = (bf16_t)(k2 * c + k1 * s);
}

// ---------------------------------------------------------------------------
// 4) Flash attention with causal mask + T5 rel-pos bias.
//    Block = 4 waves (128 thr) = 64 query rows of one (b,h); wave = 16 rows.
//    Key blocks of 32; online softmax; P re-layouted through LDS.
//    V-fragment loads issued before the softmax VALU block to hide latency.
// ---------------------------------------------------------------------------
__global__ __launch_bounds__(128) void k_attn(
    const bf16_t* __restrict__ Qr, const bf16_t* __restrict__ Kr,
    const bf16_t* __restrict__ Vt, const float* __restrict__ rel_emb,
    bf16_t* __restrict__ Y) {
  constexpr int T = 2048, D = 128, H = 16;
  constexpr float SCALE = 0.08838834764831845f;   // 1/sqrt(128)
  __shared__ float  bias_s[T];
  __shared__ bf16_t p_lds[4][16 * 32];

  const int bh = blockIdx.y;
  const int h  = bh & (H - 1);
  const int lane = threadIdx.x & 31;
  const int wid  = threadIdx.x >> 5;

  // bias table: dist = q-k >= 0 -> bucket -> rel_emb[bucket, h]
  for (int n = (int)threadIdx.x; n < T; n += 128) {
    int bucket;
    if (n < 16) bucket = n;
    else {
      int vl = 16 + (int)(__logf((float)n * (1.f / 16.f)) * (16.f / 2.0794415416798357f)); // ln(8)
      bucket = vl < 31 ? vl : 31;
    }
    bias_s[n] = rel_emb[bucket * H + h];
  }
  __syncthreads();

  const int t0 = blockIdx.x * 64 + wid * 16;
  const size_t hb = (size_t)bh * T * D;

  v16bf qa[4];
#pragma unroll
  for (int ci = 0; ci < 4; ++ci)
    qa[ci] = load_a_frag(Qr + hb + (size_t)t0 * D + 32 * ci, D, lane);

  v8f o[8] = {};
  float mrow[8], lrow[8];
#pragma unroll
  for (int r = 0; r < 8; ++r) { mrow[r] = -1e30f; lrow[r] = 0.f; }

  const int half = lane >> 4, l = lane & 15;
  const int rowoff = half * 8;
  bf16_t* pw = &p_lds[wid][0];

  const int nkb = (t0 + 47) >> 5;   // key blocks of 32 covering keys <= t0+15
  for (int kb = 0; kb < nkb; ++kb) {
    const int kbase = kb * 32;

    // all K fragments up front (one load clause), then the WMMA chains
    v16bf kf[8];
#pragma unroll
    for (int ci = 0; ci < 4; ++ci) {
      kf[ci]     = load_b_frag(Kr + hb + (size_t)kbase * D + 32 * ci, D, lane);
      kf[ci + 4] = load_b_frag(Kr + hb + (size_t)(kbase + 16) * D + 32 * ci, D, lane);
    }
    v8f s0 = {}, s1 = {};
#pragma unroll
    for (int ci = 0; ci < 4; ++ci) s0 = WMMA_BF16(qa[ci], kf[ci], s0);
#pragma unroll
    for (int ci = 0; ci < 4; ++ci) s1 = WMMA_BF16(qa[ci], kf[ci + 4], s1);

    // issue V loads now; consumed after the softmax VALU block
    v16bf bv[8];
#pragma unroll
    for (int ct = 0; ct < 8; ++ct)
      bv[ct] = load_b_frag(Vt + hb + (size_t)(ct * 16) * T + kbase, T, lane);

#pragma unroll
    for (int r = 0; r < 8; ++r) {
      const int q  = t0 + rowoff + r;
      const int ka = kbase + l, kc = kbase + 16 + l;
      float v0 = (ka <= q) ? s0[r] * SCALE + bias_s[q - ka] : -1e30f;
      float v1 = (kc <= q) ? s1[r] * SCALE + bias_s[q - kc] : -1e30f;
      float cand = fmaxf(v0, v1);
      cand = fmaxf(cand, __shfl_xor(cand, 1));
      cand = fmaxf(cand, __shfl_xor(cand, 2));
      cand = fmaxf(cand, __shfl_xor(cand, 4));
      cand = fmaxf(cand, __shfl_xor(cand, 8));
      const float mnew  = fmaxf(mrow[r], cand);
      const float alpha = __expf(mrow[r] - mnew);
      const float p0 = __expf(v0 - mnew);
      const float p1 = __expf(v1 - mnew);
      float rs = p0 + p1;
      rs += __shfl_xor(rs, 1);
      rs += __shfl_xor(rs, 2);
      rs += __shfl_xor(rs, 4);
      rs += __shfl_xor(rs, 8);
      lrow[r] = lrow[r] * alpha + rs;
      mrow[r] = mnew;
#pragma unroll
      for (int ct = 0; ct < 8; ++ct) o[ct][r] *= alpha;
      pw[(rowoff + r) * 32 + l]      = (bf16_t)p0;
      pw[(rowoff + r) * 32 + 16 + l] = (bf16_t)p1;
    }
    __builtin_amdgcn_wave_barrier();          // keep DS store->load ordered
    v16bf pa = load_a_frag(pw, 32, lane);     // P as 16x32 A fragment
#pragma unroll
    for (int ct = 0; ct < 8; ++ct)
      o[ct] = WMMA_BF16(pa, bv[ct], o[ct]);
  }

  const int b = bh >> 4;
#pragma unroll
  for (int r = 0; r < 8; ++r) {
    const float inv_l = 1.f / lrow[r];
    const size_t row = ((size_t)b * T + t0 + rowoff + r) * 2048 + (size_t)h * D + l;
#pragma unroll
    for (int ct = 0; ct < 8; ++ct)
      Y[row + ct * 16] = (bf16_t)(o[ct][r] * inv_l);
  }
}

// ---------------------------------------------------------------------------
// 5) Output projection: Out[n,j] = sum_c Y[n,c] * Wo[j,c]   (f32 output)
// ---------------------------------------------------------------------------
__global__ __launch_bounds__(256) void k_gemm_out(
    const bf16_t* __restrict__ A, const bf16_t* __restrict__ W,
    float* __restrict__ Out) {
  constexpr int K = 2048, N = 2048;
  __shared__ __align__(64) bf16_t ldsB[2 * 128 * 32];

  const int lane = threadIdx.x & 31;
  const int wid  = threadIdx.x >> 5;
  const int m0 = blockIdx.y * 128 + wid * 16;
  const int n0 = blockIdx.x * 128;

  GemmAcc g = gemm_core_128(A, W, K, m0, n0, ldsB);

  const int half = lane >> 4, l = lane & 15;
#pragma unroll
  for (int r = 0; r < 8; ++r) {
    size_t row = (size_t)(m0 + half * 8 + r) * N;
#pragma unroll
    for (int ct = 0; ct < 8; ++ct)
      Out[row + n0 + ct * 16 + l] = g.acc[ct][r];
  }
}

// ---------------------------------------------------------------------------
// Launch
// ---------------------------------------------------------------------------
extern "C" void kernel_launch(void* const* d_in, const int* in_sizes, int n_in,
                              void* d_out, int out_size, void* d_ws, size_t ws_size,
                              hipStream_t stream) {
  constexpr int Bb = 2, T = 2048, C = 2048, H = 16, D = 128;
  const float* x     = (const float*)d_in[0];
  const float* w_qkv = (const float*)d_in[1];
  const float* w_o   = (const float*)d_in[2];
  const float* rel   = (const float*)d_in[3];

  char* ws = (char*)d_ws;
  size_t off = 0;
  auto alloc = [&](size_t bytes) -> void* {
    void* p = ws + off;
    off += (bytes + 255) & ~(size_t)255;
    return p;
  };
  bf16_t* xb    = (bf16_t*)alloc((size_t)Bb * T * C * 2);
  bf16_t* wqkvb = (bf16_t*)alloc((size_t)3 * C * C * 2);
  bf16_t* wob   = (bf16_t*)alloc((size_t)C * C * 2);
  bf16_t* Qr    = (bf16_t*)alloc((size_t)Bb * H * T * D * 2);
  bf16_t* Kr    = (bf16_t*)alloc((size_t)Bb * H * T * D * 2);
  bf16_t* Vt    = (bf16_t*)alloc((size_t)Bb * H * T * D * 2);
  bf16_t* Y     = (bf16_t*)alloc((size_t)Bb * T * C * 2);

  k_cvt<<<(Bb * T * C / 4 + 255) / 256, 256, 0, stream>>>(x, xb, Bb * T * C / 4);
  k_cvt<<<(3 * C * C / 4 + 255) / 256, 256, 0, stream>>>(w_qkv, wqkvb, 3 * C * C / 4);
  k_cvt<<<(C * C / 4 + 255) / 256, 256, 0, stream>>>(w_o, wob, C * C / 4);

  k_gemm_qkv<<<dim3(3 * C / 128, (Bb * T) / 128), 256, 0, stream>>>(xb, wqkvb, Qr, Kr, Vt);

  int total = Bb * H * T * (D / 2);
  k_rope<<<(total + 255) / 256, 256, 0, stream>>>(Qr, Kr, total);

  k_attn<<<dim3(T / 64, Bb * H), 128, 0, stream>>>(Qr, Kr, Vt, rel, Y);

  k_gemm_out<<<dim3(C / 128, (Bb * T) / 128), 256, 0, stream>>>(Y, wob, (float*)d_out);
}
